// PDNConv_64707977282150
// MI455X (gfx1250) — matrix operand order repaired and checked by reading, not compiled, not run on hardware
//
#include <hip/hip_runtime.h>
#include <hip/hip_bf16.h>

// ---------------------------------------------------------------------------
// PDNConv fused pipeline for gfx1250 (MI455X), wave32 + v_wmma_f32_16x16x32_f16
//
//   K0  zero        : d_out (accumulator) + cnt
//   Kp  f32->f16    : x, W_node, W_root, W_e1, W_e2  (weights stay L2-resident)
//   K1  gate_self   : sigmoid(W_e2 @ relu(b_e1) + b_e2)  (shared by all self loops)
//   K2  node_gemm   : node_out = x@W_node^T (f16), root = x@W_root^T + b_root (f32)
//   K3  edge fused  : per 16-edge tile: MLP(2 WMMA GEMMs) -> sigmoid -> gather
//                     node_out[row] -> atomic scatter-add into d_out, count deg
//   K4  finalize    : out = (accum + node_out*gate_self)/(cnt+1) + root
//
// Assumes N % 16 == 0 and E % 16 == 0 (100000 / 800000 in this problem).
// Workspace use ~257 MB.
// ---------------------------------------------------------------------------

typedef __attribute__((ext_vector_type(16))) _Float16 v16h;
typedef __attribute__((ext_vector_type(8)))  _Float16 v8h;
typedef __attribute__((ext_vector_type(8)))  float    v8f;

#define IN_C   512
#define OUT_C  256
#define EDGE_C 128
#define HID_C  256

#define WMMA_F16(a, b, c) \
  __builtin_amdgcn_wmma_f32_16x16x32_f16(false, (a), false, (b), (short)0, (c), false, false)

// B fragment (32x16 KxN) from row-major W[col][k] (== W^T of the math).
// Lane layout: col = n0 + lane%16; lanes<16: K = k0+{0..7, 16..23}; lanes>=16: +8.
static __device__ __forceinline__ v16h load_b_frag(const _Float16* w, int ld, int n0, int k0, int lane) {
  const _Float16* q = w + (size_t)(n0 + (lane & 15)) * ld + (k0 + ((lane >> 4) << 3));
  v8h lo = *(const v8h*)q;
  v8h hi = *(const v8h*)(q + 16);
  v16h r;
#pragma unroll
  for (int j = 0; j < 8; ++j) { r[j] = lo[j]; r[j + 8] = hi[j]; }
  return r;
}

// A fragment (16x32 MxK) from row-major f16 matrix, ld halves.
// Lane layout: row m0+lane%16; lanes<16: K = k0+{0..7, 16..23}; lanes>=16: +8.
static __device__ __forceinline__ v16h load_a_frag(const _Float16* m, int ld, int m0, int k0, int lane) {
  const _Float16* q = m + (size_t)(m0 + (lane & 15)) * ld + (k0 + ((lane >> 4) << 3));
  v8h lo = *(const v8h*)q;
  v8h hi = *(const v8h*)(q + 16);
  v16h r;
#pragma unroll
  for (int j = 0; j < 8; ++j) { r[j] = lo[j]; r[j + 8] = hi[j]; }
  return r;
}

// A fragment built from f32 source (edge_attr), converted on the fly.
static __device__ __forceinline__ v16h load_a_frag_f32(const float* m, int ld, int m0, int k0, int lane) {
  const float* q = m + (size_t)(m0 + (lane & 15)) * ld + (k0 + ((lane >> 4) << 3));
  float4 a0 = *(const float4*)q;
  float4 a1 = *(const float4*)(q + 4);
  float4 a2 = *(const float4*)(q + 16);
  float4 a3 = *(const float4*)(q + 20);
  v16h r;
  r[0]  = (_Float16)a0.x; r[1]  = (_Float16)a0.y; r[2]  = (_Float16)a0.z; r[3]  = (_Float16)a0.w;
  r[4]  = (_Float16)a1.x; r[5]  = (_Float16)a1.y; r[6]  = (_Float16)a1.z; r[7]  = (_Float16)a1.w;
  r[8]  = (_Float16)a2.x; r[9]  = (_Float16)a2.y; r[10] = (_Float16)a2.z; r[11] = (_Float16)a2.w;
  r[12] = (_Float16)a3.x; r[13] = (_Float16)a3.y; r[14] = (_Float16)a3.z; r[15] = (_Float16)a3.w;
  return r;
}

// ---------------------------------------------------------------------------
__global__ void zero_f32_kernel(float* p, size_t n) {
  size_t i = (size_t)blockIdx.x * blockDim.x + threadIdx.x;
  size_t s = (size_t)gridDim.x * blockDim.x;
  for (; i < n; i += s) p[i] = 0.0f;
}

__global__ void f32_to_f16_kernel(const float* src, _Float16* dst, size_t n) {
  size_t i = (size_t)blockIdx.x * blockDim.x + threadIdx.x;
  size_t s = (size_t)gridDim.x * blockDim.x;
  for (; i < n; i += s) dst[i] = (_Float16)src[i];
}

// gate_self[c] = sigmoid( sum_j W_e2[c,j]*relu(b_e1[j]) + b_e2[c] )  (1 block, 256 thr)
__global__ void gate_self_kernel(const float* __restrict__ b_e1, const float* __restrict__ W_e2,
                                 const float* __restrict__ b_e2, float* __restrict__ gate_self) {
  int c = threadIdx.x;
  float acc = b_e2[c];
  for (int j = 0; j < HID_C; ++j) {
    float h = b_e1[j];
    h = h > 0.0f ? h : 0.0f;
    acc += W_e2[c * HID_C + j] * h;
  }
  gate_self[c] = 1.0f / (1.0f + __expf(-acc));
}

// ---------------------------------------------------------------------------
// K2: per wave, one 16-row tile of x: node_out (f16) and root (f32, +b_root).
__global__ __launch_bounds__(256) void node_gemm_kernel(
    const _Float16* __restrict__ x_h, const _Float16* __restrict__ Wnh,
    const _Float16* __restrict__ Wrh, const float* __restrict__ b_root,
    _Float16* __restrict__ node_out_h, float* __restrict__ root, int nTiles) {
  const int w = threadIdx.x >> 5, lane = threadIdx.x & 31;
  for (int tile = blockIdx.x * 8 + w; tile < nTiles; tile += gridDim.x * 8) {
    const int m0 = tile << 4;
    v16h a[16];                                       // full 16x512 x-tile in regs
#pragma unroll
    for (int k = 0; k < 16; ++k) a[k] = load_a_frag(x_h, IN_C, m0, k * 32, lane);

    const int nb = lane & 15, mh = (lane >> 4) << 3;
    for (int nt = 0; nt < 16; ++nt) {                 // node_out = x @ W_node^T
      v8f acc = {0.f, 0.f, 0.f, 0.f, 0.f, 0.f, 0.f, 0.f};
#pragma unroll
      for (int k = 0; k < 16; ++k)
        acc = WMMA_F16(a[k], load_b_frag(Wnh, IN_C, nt * 16, k * 32, lane), acc);
      const int n = nt * 16 + nb;
#pragma unroll
      for (int r = 0; r < 8; ++r)
        node_out_h[(size_t)(m0 + mh + r) * OUT_C + n] = (_Float16)acc[r];
    }
    for (int nt = 0; nt < 16; ++nt) {                 // root = x @ W_root^T + b_root
      v8f acc = {0.f, 0.f, 0.f, 0.f, 0.f, 0.f, 0.f, 0.f};
#pragma unroll
      for (int k = 0; k < 16; ++k)
        acc = WMMA_F16(a[k], load_b_frag(Wrh, IN_C, nt * 16, k * 32, lane), acc);
      const int n = nt * 16 + nb;
      const float bb = b_root[n];
#pragma unroll
      for (int r = 0; r < 8; ++r)
        root[(size_t)(m0 + mh + r) * OUT_C + n] = acc[r] + bb;
    }
  }
}

// ---------------------------------------------------------------------------
// K3: fused edge MLP + gather + gated scatter-add. One 16-edge tile per wave.
__global__ __launch_bounds__(256) void edge_fused_kernel(
    const int* __restrict__ edge_index, const float* __restrict__ edge_attr,
    const _Float16* __restrict__ We1h, const _Float16* __restrict__ We2h,
    const float* __restrict__ b_e1, const float* __restrict__ b_e2,
    const _Float16* __restrict__ node_out_h, float* __restrict__ out_accum,
    int* __restrict__ cnt, int nTiles, int E) {
  __shared__ _Float16 h_sh[8][16 * HID_C];            // 8 KB per wave, 64 KB total
  const int w = threadIdx.x >> 5, lane = threadIdx.x & 31;
  _Float16* hbuf = h_sh[w];

  for (int tile = blockIdx.x * 8 + w; tile < nTiles; tile += gridDim.x * 8) {
    const int e0 = tile << 4;
    int myRow = 0, myCol = 0;
    if (lane < 16) {
      myRow = edge_index[e0 + lane];                  // source node (row)
      myCol = edge_index[E + e0 + lane];              // destination node (col)
    }
    const int nb = lane & 15, mh = (lane >> 4) << 3;

    // ---- layer 1: h = relu(ea @ W_e1^T + b_e1), K=128 -> LDS (f16, row-major 16x256)
    v16h a1[4];
#pragma unroll
    for (int k = 0; k < 4; ++k) a1[k] = load_a_frag_f32(edge_attr, EDGE_C, e0, k * 32, lane);
    for (int nt = 0; nt < 16; ++nt) {
      const int n = nt * 16 + nb;
      const float b1 = b_e1[n];
      v8f acc = {b1, b1, b1, b1, b1, b1, b1, b1};
#pragma unroll
      for (int k = 0; k < 4; ++k)
        acc = WMMA_F16(a1[k], load_b_frag(We1h, EDGE_C, nt * 16, k * 32, lane), acc);
#pragma unroll
      for (int r = 0; r < 8; ++r) {
        float v = acc[r];
        hbuf[(mh + r) * HID_C + n] = (_Float16)(v > 0.0f ? v : 0.0f);
      }
    }
    asm volatile("" ::: "memory");                    // keep DS stores before DS loads
    // (same-wave LDS ops are processed in order per CDNA5 ISA; no barrier needed)

    // ---- layer 2: gate = sigmoid(h @ W_e2^T + b_e2), fused gather+scatter
    v16h a2[8];
#pragma unroll
    for (int k = 0; k < 8; ++k) a2[k] = load_a_frag(hbuf, HID_C, 0, k * 32, lane);
    for (int nt = 0; nt < 16; ++nt) {
      const int n = nt * 16 + nb;
      const float b2 = b_e2[n];
      v8f acc = {b2, b2, b2, b2, b2, b2, b2, b2};
#pragma unroll
      for (int k = 0; k < 8; ++k)
        acc = WMMA_F16(a2[k], load_b_frag(We2h, HID_C, nt * 16, k * 32, lane), acc);
#pragma unroll
      for (int r = 0; r < 8; ++r) {
        const int m = mh + r;                         // local edge index 0..15
        const float gate = 1.0f / (1.0f + __expf(-acc[r]));
        const int row = __shfl(myRow, m, 32);
        const int col = __shfl(myCol, m, 32);
        const float nv = (float)node_out_h[(size_t)row * OUT_C + n];
        atomicAdd(&out_accum[(size_t)col * OUT_C + n], nv * gate);
      }
    }
    if (lane < 16) atomicAdd(&cnt[myCol], 1);         // in-degree (real edges)
  }
}

// ---------------------------------------------------------------------------
// K4: out = (accum + node_out*gate_self) / (cnt+1) + root
__global__ void finalize_kernel(float* __restrict__ out, const _Float16* __restrict__ node_out_h,
                                const float* __restrict__ root, const float* __restrict__ gate_self,
                                const int* __restrict__ cnt, size_t total) {
  size_t i = (size_t)blockIdx.x * blockDim.x + threadIdx.x;
  size_t s = (size_t)gridDim.x * blockDim.x;
  for (; i < total; i += s) {
    const int node = (int)(i >> 8);
    const int c = (int)(i & 255);
    const float deg = (float)(cnt[node] + 1);         // +1 self loop, always >= 1
    out[i] = (out[i] + (float)node_out_h[i] * gate_self[c]) / deg + root[i];
  }
}

// ---------------------------------------------------------------------------
static inline int nblk(size_t n, int per, int cap) {
  size_t b = (n + (size_t)per - 1) / (size_t)per;
  if (b > (size_t)cap) b = (size_t)cap;
  return b < 1 ? 1 : (int)b;
}

extern "C" void kernel_launch(void* const* d_in, const int* in_sizes, int n_in,
                              void* d_out, int out_size, void* d_ws, size_t ws_size,
                              hipStream_t stream) {
  // setup_inputs order: x, edge_index, edge_attr, W_node, W_e1, b_e1, W_e2, b_e2, W_root, b_root
  const float* x          = (const float*)d_in[0];
  const int*   edge_index = (const int*)d_in[1];     // harness integer convention
  const float* edge_attr  = (const float*)d_in[2];
  const float* W_node     = (const float*)d_in[3];
  const float* W_e1       = (const float*)d_in[4];
  const float* b_e1       = (const float*)d_in[5];
  const float* W_e2       = (const float*)d_in[6];
  const float* b_e2       = (const float*)d_in[7];
  const float* W_root     = (const float*)d_in[8];
  const float* b_root     = (const float*)d_in[9];
  float* out = (float*)d_out;

  const int N = in_sizes[0] / IN_C;                  // 100000
  const int E = in_sizes[2] / EDGE_C;                // 800000

  // -------- workspace carve-out (256B aligned), ~257 MB total --------
  char* ws = (char*)d_ws;
  size_t off = 0;
  auto carve = [&](size_t bytes) -> void* {
    void* p = ws + off;
    off += (bytes + 255) & ~(size_t)255;
    return p;
  };
  _Float16* x_h        = (_Float16*)carve((size_t)N * IN_C * 2);
  _Float16* node_out_h = (_Float16*)carve((size_t)N * OUT_C * 2);
  float*    root       = (float*)   carve((size_t)N * OUT_C * 4);
  _Float16* Wnh        = (_Float16*)carve((size_t)OUT_C * IN_C * 2);
  _Float16* Wrh        = (_Float16*)carve((size_t)OUT_C * IN_C * 2);
  _Float16* We1h       = (_Float16*)carve((size_t)HID_C * EDGE_C * 2);
  _Float16* We2h       = (_Float16*)carve((size_t)OUT_C * HID_C * 2);
  float*    gate_self  = (float*)   carve(OUT_C * 4);
  int*      cnt        = (int*)     carve((size_t)N * 4);
  (void)ws_size; (void)n_in; (void)out_size;

  const size_t nOut = (size_t)N * OUT_C;

  // K0: zero accumulator (d_out) and degree counters
  zero_f32_kernel<<<nblk(nOut, 256, 8192), 256, 0, stream>>>(out, nOut);
  zero_f32_kernel<<<nblk((size_t)N, 256, 1024), 256, 0, stream>>>((float*)cnt, (size_t)N);

  // Kp: f32 -> f16 conversions (x + the four weight matrices)
  f32_to_f16_kernel<<<nblk((size_t)N * IN_C, 256, 8192), 256, 0, stream>>>(x, x_h, (size_t)N * IN_C);
  f32_to_f16_kernel<<<nblk((size_t)OUT_C * IN_C, 256, 1024), 256, 0, stream>>>(W_node, Wnh, (size_t)OUT_C * IN_C);
  f32_to_f16_kernel<<<nblk((size_t)OUT_C * IN_C, 256, 1024), 256, 0, stream>>>(W_root, Wrh, (size_t)OUT_C * IN_C);
  f32_to_f16_kernel<<<nblk((size_t)HID_C * EDGE_C, 256, 1024), 256, 0, stream>>>(W_e1, We1h, (size_t)HID_C * EDGE_C);
  f32_to_f16_kernel<<<nblk((size_t)OUT_C * HID_C, 256, 1024), 256, 0, stream>>>(W_e2, We2h, (size_t)OUT_C * HID_C);

  // K1: shared self-loop gate vector
  gate_self_kernel<<<1, 256, 0, stream>>>(b_e1, W_e2, b_e2, gate_self);

  // K2: node GEMMs (16 rows per wave, 8 waves per block)
  const int nTilesN = N >> 4;
  node_gemm_kernel<<<nblk((size_t)nTilesN, 8, 8192), 256, 0, stream>>>(
      x_h, Wnh, Wrh, b_root, node_out_h, root, nTilesN);

  // K3: fused edge MLP + gather + scatter (16 edges per wave)
  const int nTilesE = E >> 4;
  edge_fused_kernel<<<nblk((size_t)nTilesE, 8, 8192), 256, 0, stream>>>(
      edge_index, edge_attr, We1h, We2h, b_e1, b_e2, node_out_h, out, cnt, nTilesE, E);

  // K4: normalize + root term
  finalize_kernel<<<nblk(nOut, 256, 8192), 256, 0, stream>>>(
      out, node_out_h, root, gate_self, cnt, nOut);
}
